// QuantumAttentionLayer_9113920602186
// MI455X (gfx1250) — matrix operand order repaired and checked by reading, not compiled, not run on hardware
//
#include <hip/hip_runtime.h>

// ---------------- problem constants ----------------
#define LL 2048   // query length
#define SS 2048   // key length
#define NB 4      // batch
#define EE 1024   // embed dim
#define HH 16     // heads
#define DD 64     // head dim

typedef __attribute__((ext_vector_type(16))) _Float16 v16h;
typedef __attribute__((ext_vector_type(8)))  float    v8f;

#ifndef __has_builtin
#define __has_builtin(x) 0
#endif
#if __has_builtin(__builtin_amdgcn_tensor_load_to_lds) && \
    __has_builtin(__builtin_amdgcn_s_wait_tensorcnt)
#define USE_TDM 1
#else
#define USE_TDM 0
#endif

// ---------------- WMMA fragment helpers (CDNA5 16x16x32 f16, wave32) -------
// A matrix 16x32 (MxK): lanes 0-15 -> M=lane, K = klo..klo+7 (v0..v3) and
// klo+16..klo+23 (v4..v7) with klo = 0; lanes 16-31 -> klo = 8.
__device__ __forceinline__ v16h load_a_frag(const _Float16* __restrict__ p,
                                            int ld, int rbase, int kbase, int lane) {
    const int m   = lane & 15;
    const int klo = (lane >> 4) * 8;
    const _Float16* r = p + (size_t)(rbase + m) * ld + kbase + klo;
    v16h a;
#pragma unroll
    for (int i = 0; i < 8; ++i) { a[i] = r[i]; a[8 + i] = r[16 + i]; }
    return a;
}

// B matrix 32x16 (KxN) with B[k][n] = M2[n][k] for row-major M2 (ld elems/row):
// each lane reads K = klo..klo+7 and klo+16..klo+23 (klo = 8*(lane>>4)) from row (nbase + lane&15).
__device__ __forceinline__ v16h load_bT_frag(const _Float16* __restrict__ p,
                                             int ld, int nbase, int kbase, int lane) {
    const int n     = lane & 15;
    const int khalf = (lane >> 4) * 8;
    const _Float16* r = p + (size_t)(nbase + n) * ld + kbase + khalf;
    v16h b;
#pragma unroll
    for (int i = 0; i < 8; ++i) { b[i] = r[i]; b[8 + i] = r[16 + i]; }
    return b;
}

__device__ __forceinline__ v8f wmma_f16(v16h a, v16h b, v8f c) {
    return __builtin_amdgcn_wmma_f32_16x16x32_f16(false, a, false, b,
                                                  (short)0, c, false, false);
}

// ---------------- Tensor Data Mover: 2D tile global -> LDS ----------------
#if USE_TDM
typedef unsigned int v4u __attribute__((ext_vector_type(4)));
typedef int          v4i __attribute__((ext_vector_type(4)));
typedef int          v8i __attribute__((ext_vector_type(8)));

// Load a tile0 x tile1 (f16) tile from a row-major tensor (dim0 elems/row,
// stride0 row stride) into LDS at byte offset lds_off, inserting a 4-DWORD
// (16B) pad after every 32 DWORDs (128B = one 64-f16 row).
__device__ __forceinline__ void tdm_load_2d_f16(unsigned lds_off,
                                                const _Float16* gptr,
                                                unsigned dim0, unsigned dim1,
                                                unsigned tile0, unsigned tile1,
                                                unsigned stride0) {
    unsigned long long ga = (unsigned long long)(__UINTPTR_TYPE__)gptr;
    v4u g0;
    g0[0] = 1u;                                   // count=1, user descriptor
    g0[1] = lds_off;                              // LDS byte address
    g0[2] = (unsigned)ga;                         // global addr lo
    g0[3] = (unsigned)(ga >> 32) | (2u << 30);    // global addr hi | type=2
    v8i g1;
    g1[0] = (1 << 16)                             // data_size = 2 bytes
          | (1 << 20)                             // pad_enable
          | (4 << 22)                             // pad_interval: 32 DWORDs
          | (3 << 25);                            // pad_amount: 4 DWORDs (16B)
    g1[1] = (int)((dim0 & 0xffffu) << 16);                        // tensor_dim0 lo
    g1[2] = (int)(((dim0 >> 16) & 0xffffu) | ((dim1 & 0xffffu) << 16));
    g1[3] = (int)(((dim1 >> 16) & 0xffffu) | ((tile0 & 0xffffu) << 16));
    g1[4] = (int)(tile1 & 0xffffu);               // tile_dim1 (tile_dim2 = 0)
    g1[5] = (int)stride0;                         // tensor_dim0_stride lo
    g1[6] = 0;
    g1[7] = 0;
    v4i z4 = {0, 0, 0, 0};
#if __clang_major__ >= 23
    v8i z8 = {0, 0, 0, 0, 0, 0, 0, 0};
    __builtin_amdgcn_tensor_load_to_lds(g0, g1, z4, z4, z8, 0);
#else
    __builtin_amdgcn_tensor_load_to_lds(g0, g1, z4, z4, 0);
#endif
}
#endif

// ---------------- f32 -> f16 conversion -----------------------------------
__global__ void cvt_f32_f16(const float* __restrict__ in,
                            _Float16* __restrict__ out, int n) {
    int i = blockIdx.x * blockDim.x + threadIdx.x;
    int stride = gridDim.x * blockDim.x;
    for (; i < n; i += stride) out[i] = (_Float16)in[i];
}

// ---------------- projection GEMM: Y = X @ W^T + b ------------------------
// Register-blocked: each wave computes a 32(M) x 64(N) output region
// (2 A-frags + 4 B-frags -> 8 WMMAs per 32-wide k step; ~21 FLOP/B).
// MODE 0: f16 -> Qp [n,h,L,D]; 1: Kp [n,h,S,D]; 2: VpT [n,h,D,S];
// MODE 3: f32 -> out [M,E].
template <int MODE>
__global__ void __launch_bounds__(256)
proj_kernel(const _Float16* __restrict__ X, const _Float16* __restrict__ W,
            const float* __restrict__ bias, _Float16* __restrict__ outH,
            float* __restrict__ outF) {
    const int lane   = threadIdx.x & 31;
    const int wid    = threadIdx.x >> 5;
    const int region = blockIdx.x * 8 + wid;  // (8192/32)*(1024/64) = 4096
    const int rn     = region & 15;           // N region: 0..15 (64 cols each)
    const int rm     = region >> 4;           // M region: 0..255 (32 rows each)

    v8f acc[2][4];
#pragma unroll
    for (int mi = 0; mi < 2; ++mi)
#pragma unroll
        for (int ni = 0; ni < 4; ++ni) acc[mi][ni] = (v8f){};

    for (int kb = 0; kb < EE; kb += 32) {
        if (kb + 64 <= EE) {
            __builtin_prefetch((const void*)(W + (size_t)(rn * 64 + (lane & 15)) * EE + kb + 32), 0, 1);
            __builtin_prefetch((const void*)(X + (size_t)(rm * 32 + (lane & 15)) * EE + kb + 32), 0, 1);
        }
        v16h a0 = load_a_frag(X, EE, rm * 32,      kb, lane);
        v16h a1 = load_a_frag(X, EE, rm * 32 + 16, kb, lane);
        v16h b[4];
#pragma unroll
        for (int ni = 0; ni < 4; ++ni)
            b[ni] = load_bT_frag(W, EE, rn * 64 + ni * 16, kb, lane);
#pragma unroll
        for (int ni = 0; ni < 4; ++ni) {
            acc[0][ni] = wmma_f16(a0, b[ni], acc[0][ni]);
            acc[1][ni] = wmma_f16(a1, b[ni], acc[1][ni]);
        }
    }

    const int colN   = lane & 15;
    const int rowoff = (lane >> 4) * 8;
#pragma unroll
    for (int mi = 0; mi < 2; ++mi) {
#pragma unroll
        for (int ni = 0; ni < 4; ++ni) {
            const int nout = rn * 64 + ni * 16 + colN;
            const float bv = bias[nout];
#pragma unroll
            for (int i = 0; i < 8; ++i) {
                const int   mrow = rm * 32 + mi * 16 + i + rowoff;
                const float val  = acc[mi][ni][i] + bv;
                if (MODE == 3) {
                    outF[(size_t)mrow * EE + nout] = val;
                } else {
                    const int pos = mrow >> 2;   // l or s (rows are pos*NB+nb)
                    const int nb  = mrow & 3;
                    const int hh  = nout >> 6;
                    const int dd  = nout & 63;
                    size_t idx;
                    if (MODE == 0)      idx = ((size_t)(nb * HH + hh) * LL + pos) * DD + dd;
                    else if (MODE == 1) idx = ((size_t)(nb * HH + hh) * SS + pos) * DD + dd;
                    else                idx = ((size_t)(nb * HH + hh) * DD + dd) * SS + pos;
                    outH[idx] = (_Float16)val;
                }
            }
        }
    }
}

// ---------------- flash-style attention -----------------------------------
// One wave per (n,h, 16-row Q tile); 4 waves/block share one (n,h) head, so
// K / V^T panels are staged per-block into LDS in 64x64 chunks (TDM when
// available, cooperative loads otherwise).  LDS rows padded to 72 f16
// (144B = 36 banks) for conflict-free ds_load_b128 fragments.
#define CH_ROWS 64
#define LDPAD   72

__global__ void __launch_bounds__(128)
attn_kernel(const _Float16* __restrict__ Qp, const _Float16* __restrict__ Kp,
            const _Float16* __restrict__ VpT, _Float16* __restrict__ Ctx,
            float* __restrict__ Wmean) {
    __shared__ _Float16 Ksh[CH_ROWS][LDPAD];   // K rows (s_local) x d
    __shared__ _Float16 Vsh[CH_ROWS][LDPAD];   // V^T rows (d) x s_local
    __shared__ _Float16 pbuf[4][16][40];       // per-wave 16x32 P staging

    const int lane = threadIdx.x & 31;
    const int wid  = threadIdx.x >> 5;
    const int tid  = threadIdx.x;
    const int gid  = blockIdx.x * 4 + wid;
    const int lt   = gid & (LL / 16 - 1);
    const int h    = (gid >> 7) & (HH - 1);
    const int nb   = gid >> 11;

    const _Float16* qb = Qp  + (size_t)(nb * HH + h) * LL * DD;
    const _Float16* kb = Kp  + (size_t)(nb * HH + h) * SS * DD;
    const _Float16* vt = VpT + (size_t)(nb * HH + h) * DD * SS;

    const v16h aq0 = load_a_frag(qb, DD, lt * 16, 0,  lane);
    const v16h aq1 = load_a_frag(qb, DD, lt * 16, 32, lane);

    const float scale = 0.125f;                  // 1/sqrt(64)
    float rm[8], rl[8];
#pragma unroll
    for (int i = 0; i < 8; ++i) { rm[i] = -3.0e38f; rl[i] = 0.0f; }

    // ---- pass 1: online row max / sum over all S tiles ------------------
    for (int ch = 0; ch < SS / CH_ROWS; ++ch) {
#if USE_TDM
        if (wid == 0) {
            tdm_load_2d_f16((unsigned)(__UINTPTR_TYPE__)&Ksh[0][0],
                            kb + (size_t)ch * CH_ROWS * DD,
                            DD, SS, DD, CH_ROWS, DD);
            __builtin_amdgcn_s_wait_tensorcnt(0);
        }
#else
        // cooperative: 64x64 f16 = 512 16B chunks, 4 per thread
        for (int j = 0; j < 4; ++j) {
            const int cid = tid + 128 * j;
            const int r = cid >> 3, c8 = (cid & 7) * 8;
            const _Float16* src = kb + (size_t)(ch * CH_ROWS + r) * DD + c8;
#pragma unroll
            for (int e = 0; e < 8; ++e) Ksh[r][c8 + e] = src[e];
        }
#endif
        __syncthreads();
#pragma unroll
        for (int t = 0; t < 4; ++t) {
            v8f c = {};
            c = wmma_f16(aq0, load_bT_frag(&Ksh[0][0], LDPAD, t * 16, 0,  lane), c);
            c = wmma_f16(aq1, load_bT_frag(&Ksh[0][0], LDPAD, t * 16, 32, lane), c);
#pragma unroll
            for (int i = 0; i < 8; ++i) {
                float v = c[i] * scale;
                float tm = v;
                tm = fmaxf(tm, __shfl_xor(tm, 1));
                tm = fmaxf(tm, __shfl_xor(tm, 2));
                tm = fmaxf(tm, __shfl_xor(tm, 4));
                tm = fmaxf(tm, __shfl_xor(tm, 8));
                const float mnew = fmaxf(rm[i], tm);
                float e = __expf(v - mnew);
                e += __shfl_xor(e, 1);
                e += __shfl_xor(e, 2);
                e += __shfl_xor(e, 4);
                e += __shfl_xor(e, 8);
                rl[i] = rl[i] * __expf(rm[i] - mnew) + e;
                rm[i] = mnew;
            }
        }
        __syncthreads();
    }
    float rinv[8];
#pragma unroll
    for (int i = 0; i < 8; ++i) rinv[i] = 1.0f / rl[i];

    // ---- pass 2: probs -> mean-weights atomics + P@V --------------------
    const int colN   = lane & 15;
    const int rowoff = (lane >> 4) * 8;
    float* wrow = Wmean + ((size_t)nb * LL + lt * 16) * SS;
    v8f oc0 = {}, oc1 = {}, oc2 = {}, oc3 = {};

    for (int ch = 0; ch < SS / CH_ROWS; ++ch) {
#if USE_TDM
        if (wid == 0) {
            tdm_load_2d_f16((unsigned)(__UINTPTR_TYPE__)&Ksh[0][0],
                            kb + (size_t)ch * CH_ROWS * DD,
                            DD, SS, DD, CH_ROWS, DD);
            tdm_load_2d_f16((unsigned)(__UINTPTR_TYPE__)&Vsh[0][0],
                            vt + (size_t)ch * CH_ROWS,
                            SS, DD, CH_ROWS, DD, SS);
            __builtin_amdgcn_s_wait_tensorcnt(0);
        }
#else
        for (int j = 0; j < 4; ++j) {
            const int cid = tid + 128 * j;
            const int r = cid >> 3, c8 = (cid & 7) * 8;
            const _Float16* srck = kb + (size_t)(ch * CH_ROWS + r) * DD + c8;
            const _Float16* srcv = vt + (size_t)r * SS + ch * CH_ROWS + c8;
#pragma unroll
            for (int e = 0; e < 8; ++e) { Ksh[r][c8 + e] = srck[e]; Vsh[r][c8 + e] = srcv[e]; }
        }
#endif
        __syncthreads();
#pragma unroll
        for (int t = 0; t < 4; ++t) {
            const int sb = ch * 4 + t;
            v8f c = {};
            c = wmma_f16(aq0, load_bT_frag(&Ksh[0][0], LDPAD, t * 16, 0,  lane), c);
            c = wmma_f16(aq1, load_bT_frag(&Ksh[0][0], LDPAD, t * 16, 32, lane), c);
#pragma unroll
            for (int i = 0; i < 8; ++i) {
                const float p = __expf(c[i] * scale - rm[i]) * rinv[i];
                atomicAdd(&wrow[(size_t)(i + rowoff) * SS + sb * 16 + colN],
                          p * (1.0f / HH));
                pbuf[wid][i + rowoff][(t & 1) * 16 + colN] = (_Float16)p;
            }
            if (t & 1) {
                asm volatile("s_wait_dscnt 0" ::: "memory");   // P tile visible
                v16h ap;
                const int m   = lane & 15;
                const int klo = (lane >> 4) * 8;
#pragma unroll
                for (int i = 0; i < 8; ++i) {
                    ap[i]     = pbuf[wid][m][klo + i];
                    ap[8 + i] = pbuf[wid][m][klo + 16 + i];
                }
                const int kloc = (t - 1) * 16;    // 32-wide k block in chunk
                oc0 = wmma_f16(ap, load_bT_frag(&Vsh[0][0], LDPAD,  0, kloc, lane), oc0);
                oc1 = wmma_f16(ap, load_bT_frag(&Vsh[0][0], LDPAD, 16, kloc, lane), oc1);
                oc2 = wmma_f16(ap, load_bT_frag(&Vsh[0][0], LDPAD, 32, kloc, lane), oc2);
                oc3 = wmma_f16(ap, load_bT_frag(&Vsh[0][0], LDPAD, 48, kloc, lane), oc3);
            }
        }
        __syncthreads();
    }

    // ---- store context tile (rows l, cols h*64 + d) as f16 --------------
#pragma unroll
    for (int i = 0; i < 8; ++i) {
        const int l = lt * 16 + i + rowoff;
        const size_t base = ((size_t)l * NB + nb) * EE + h * DD;
        Ctx[base +  0 + colN] = (_Float16)oc0[i];
        Ctx[base + 16 + colN] = (_Float16)oc1[i];
        Ctx[base + 32 + colN] = (_Float16)oc2[i];
        Ctx[base + 48 + colN] = (_Float16)oc3[i];
    }
}

// ---------------- host-side orchestration ---------------------------------
extern "C" void kernel_launch(void* const* d_in, const int* in_sizes, int n_in,
                              void* d_out, int out_size, void* d_ws, size_t ws_size,
                              hipStream_t stream) {
    const float* q  = (const float*)d_in[0];
    const float* k  = (const float*)d_in[1];
    const float* v  = (const float*)d_in[2];
    const float* Wq = (const float*)d_in[3];
    const float* bq = (const float*)d_in[4];
    const float* Wk = (const float*)d_in[5];
    const float* bk = (const float*)d_in[6];
    const float* Wv = (const float*)d_in[7];
    const float* bv = (const float*)d_in[8];
    const float* Wo = (const float*)d_in[9];
    const float* bo = (const float*)d_in[10];

    float* out      = (float*)d_out;
    float* attn_out = out;                                 // [L,N,E]
    float* wmean    = out + (size_t)LL * NB * EE;          // [N,L,S]

    const size_t XN = (size_t)LL * NB * EE;   // 8388608 f16 elems per act
    const size_t WN = (size_t)EE * EE;        // 1048576 f16 elems per weight
    _Float16* p   = (_Float16*)d_ws;
    _Float16* Xq  = p; p += XN;
    _Float16* Xk  = p; p += XN;
    _Float16* Xv  = p; p += XN;
    _Float16* Wqh = p; p += WN;
    _Float16* Wkh = p; p += WN;
    _Float16* Wvh = p; p += WN;
    _Float16* Woh = p; p += WN;
    _Float16* Qp  = p; p += XN;
    _Float16* Kp  = p; p += XN;
    _Float16* VpT = p; p += XN;
    _Float16* Ctx = p; p += XN;

    // zero the mean-weights region (accumulated with atomics over heads)
    hipMemsetAsync(wmean, 0, (size_t)NB * LL * SS * sizeof(float), stream);

    // f32 -> f16 conversions
    cvt_f32_f16<<<(int)((XN + 1023) / 1024), 256, 0, stream>>>(q, Xq, (int)XN);
    cvt_f32_f16<<<(int)((XN + 1023) / 1024), 256, 0, stream>>>(k, Xk, (int)XN);
    cvt_f32_f16<<<(int)((XN + 1023) / 1024), 256, 0, stream>>>(v, Xv, (int)XN);
    cvt_f32_f16<<<(int)((WN + 1023) / 1024), 256, 0, stream>>>(Wq, Wqh, (int)WN);
    cvt_f32_f16<<<(int)((WN + 1023) / 1024), 256, 0, stream>>>(Wk, Wkh, (int)WN);
    cvt_f32_f16<<<(int)((WN + 1023) / 1024), 256, 0, stream>>>(Wv, Wvh, (int)WN);
    cvt_f32_f16<<<(int)((WN + 1023) / 1024), 256, 0, stream>>>(Wo, Woh, (int)WN);

    // projections: (8192/32)*(1024/64) = 4096 wave regions, 8 waves/block
    const int blocks = 4096 / 8;   // 512
    proj_kernel<0><<<blocks, 256, 0, stream>>>(Xq, Wqh, bq, Qp,  nullptr);
    proj_kernel<1><<<blocks, 256, 0, stream>>>(Xk, Wkh, bk, Kp,  nullptr);
    proj_kernel<2><<<blocks, 256, 0, stream>>>(Xv, Wvh, bv, VpT, nullptr);

    // attention: 4*16*128 = 8192 waves, 4 waves (one head) per block
    attn_kernel<<<(NB * HH * (LL / 16)) / 4, 128, 0, stream>>>(Qp, Kp, VpT, Ctx, wmean);

    // output projection, f32 straight into d_out
    proj_kernel<3><<<blocks, 256, 0, stream>>>(Ctx, Woh, bo, nullptr, attn_out);
}
